// RNNWrapper_8005819040472
// MI455X (gfx1250) — matrix operand order repaired and checked by reading, not compiled, not run on hardware
//
#include <hip/hip_runtime.h>
#include <math.h>

// ---------------------------------------------------------------------------
// Types for WMMA fragments (gfx1250, wave32)
// ---------------------------------------------------------------------------
typedef __attribute__((ext_vector_type(16))) __bf16          v16bf;
typedef __attribute__((ext_vector_type(8)))  float           v8f;
typedef __attribute__((ext_vector_type(8)))  unsigned short  us8;
typedef __attribute__((ext_vector_type(16))) unsigned short  us16;

// f32 -> bf16 (round to nearest even), pure integer ops
static __device__ __forceinline__ unsigned short f2bf(float f) {
    union { float f; unsigned int u; } c; c.f = f;
    unsigned int u = c.u;
    unsigned int r = u + 0x7FFFu + ((u >> 16) & 1u);
    return (unsigned short)(r >> 16);
}

static __device__ __forceinline__ v16bf mk_frag(us8 lo, us8 hi) {
    union { struct { us8 a, b; } s; v16bf v; } u;
    u.s.a = lo; u.s.b = hi;
    return u.v;
}

// 8 bf16 at p and 8 bf16 at p+16 -> fragment
static __device__ __forceinline__ v16bf load_b_frag(const unsigned short* p) {
    return mk_frag(*reinterpret_cast<const us8*>(p),
                   *reinterpret_cast<const us8*>(p + 16));
}

static __device__ __forceinline__ v8f wmma_bf16(v16bf a, v16bf b, v8f c) {
    return __builtin_amdgcn_wmma_f32_16x16x32_bf16(
        /*neg_a=*/false, a, /*neg_b=*/false, b,
        /*c_mod=*/(short)0, c, /*reuse_a=*/false, /*reuse_b=*/false);
}

// async copy of 16 bytes global -> LDS (tracked by ASYNCcnt)
static __device__ __forceinline__ void async_cp16(unsigned lds_off,
                                                  const void* gptr) {
    unsigned long long ga = (unsigned long long)gptr;
    asm volatile("global_load_async_to_lds_b128 %0, %1, off"
                 :: "v"(lds_off), "v"(ga) : "memory");
}
static __device__ __forceinline__ void wait_async0() {
    asm volatile("s_wait_asynccnt 0x0" ::: "memory");
}
static __device__ __forceinline__ unsigned lds_off32(const void* p) {
    return (unsigned)(unsigned long long)p;   // low 32b of LDS flat addr = offset
}

// ---------------------------------------------------------------------------
// Problem constants
// ---------------------------------------------------------------------------
#define EMBED   512
#define HIDDEN  1024
#define OUTV    32000
#define BATCH   8
#define SEQ     256
#define MROWS   (SEQ * BATCH)   // 2048, row m = t*8 + b

#define HROW 1032   // rnn LDS row stride (ushorts): bank start 4*q, no conflicts
#define PROW 40     // panel LDS row stride (ushorts): 32 data + 8 pad

// ---------------------------------------------------------------------------
// Kernel A: f32 -> bf16 bulk convert (vectorized x4)
// ---------------------------------------------------------------------------
__global__ void cvt_f32_bf16(const float* __restrict__ in,
                             unsigned short* __restrict__ out, int n) {
    int i = (blockIdx.x * blockDim.x + threadIdx.x) * 4;
    if (i < n) {
        float4 v = *reinterpret_cast<const float4*>(in + i);
        union { unsigned short s[4]; unsigned long long u; } p;
        p.s[0] = f2bf(v.x); p.s[1] = f2bf(v.y);
        p.s[2] = f2bf(v.z); p.s[3] = f2bf(v.w);
        *reinterpret_cast<unsigned long long*>(out + i) = p.u;
    }
}

// ---------------------------------------------------------------------------
// Kernel B: xproj[m][n] = sum_k embed[idx[m]][k] * W_ih[n][k] + b_ih[n]
//   M = 2048 (m = t*8+b), K = 512, N = 1024.
//   Grid: (8, 128), block 256 (8 waves). One 16-row A tile per block, staged
//   to LDS (f32 gather + cvt done once, shared by all 8 waves).
// ---------------------------------------------------------------------------
__global__ __launch_bounds__(256)
void xproj_kernel(const long long* __restrict__ inputs,
                  const float* __restrict__ embed,
                  const unsigned short* __restrict__ wih_bf,
                  const float* __restrict__ b_ih,
                  float* __restrict__ xproj) {
    __shared__ __align__(16) unsigned short ldsX[16 * PROW];
    __shared__ int sRow[16];

    const int tid  = threadIdx.x;
    const int lane = tid & 31;
    const int wave = tid >> 5;
    const int half = lane >> 4;
    const int q    = lane & 15;

    const int mBase = blockIdx.y * 16;
    const int nBase = blockIdx.x * 128 + wave * 16;

    if (tid < 16) {
        const int m = mBase + tid;
        sRow[tid] = (int)inputs[(m & 7) * SEQ + (m >> 3)];
    }
    __syncthreads();

    const int srow = tid >> 4;          // 0..15  (staging row)
    const int scol = (tid & 15) * 2;    // 0..30  (staging col pair)
    const unsigned short* brow = wih_bf + (nBase + q) * EMBED;

    v8f acc = {};
    for (int kb = 0; kb < EMBED; kb += 32) {
        // stage A tile chunk: 16 rows x 32 k, two f32->bf16 per thread
        float2 v = *reinterpret_cast<const float2*>(
            embed + sRow[srow] * EMBED + kb + scol);
        unsigned packed = (unsigned)f2bf(v.x) | ((unsigned)f2bf(v.y) << 16);
        *reinterpret_cast<unsigned*>(&ldsX[srow * PROW + scol]) = packed;
        __syncthreads();

        us8 a0 = *reinterpret_cast<const us8*>(&ldsX[q * PROW + half * 8]);
        us8 a1 = *reinterpret_cast<const us8*>(&ldsX[q * PROW + half * 8 + 16]);
        v16bf af = mk_frag(a0, a1);
        v16bf bf = load_b_frag(brow + kb + half * 8);
        acc = wmma_bf16(af, bf, acc);
        __syncthreads();
    }

    const float bias = b_ih[nBase + q];
    #pragma unroll
    for (int r = 0; r < 8; ++r) {
        const int mo = mBase + half * 8 + r;
        xproj[mo * HIDDEN + nBase + q] = acc[r] + bias;
    }
}

// ---------------------------------------------------------------------------
// Kernel C: recurrence. Single workgroup of 1024 threads (32 waves).
//   Per step: h_new = tanh(xproj[t] + h @ W_hh^T + b_hh); h kept in LDS bf16.
//   Wave w covers columns [w*32, w*32+32) as two 16-wide tiles.
// ---------------------------------------------------------------------------
__global__ __launch_bounds__(1024)
void rnn_kernel(const float* __restrict__ xproj,
                const unsigned short* __restrict__ whh_bf,
                const float* __restrict__ b_hh,
                unsigned short* __restrict__ hs_bf) {
    __shared__ __align__(16) unsigned short hld[16 * HROW];

    const int tid = threadIdx.x;
    for (int i = tid; i < 16 * HROW / 8; i += 1024) {
        us8 z = {};
        reinterpret_cast<us8*>(hld)[i] = z;   // rows 8..15 stay zero (padding)
    }
    __syncthreads();

    const int lane = tid & 31;
    const int wave = tid >> 5;
    const int half = lane >> 4;
    const int q    = lane & 15;

    const int n0 = wave * 32;
    const int n1 = n0 + 16;
    const float bias0 = b_hh[n0 + q];
    const float bias1 = b_hh[n1 + q];
    const unsigned short* br0 = whh_bf + (n0 + q) * HIDDEN;
    const unsigned short* br1 = whh_bf + (n1 + q) * HIDDEN;

    for (int t = 0; t < SEQ; ++t) {
        v8f acc0 = {}, acc1 = {};
        for (int kb = 0; kb < HIDDEN; kb += 32) {
            const int klo = kb + half * 8;
            us8 a0 = *reinterpret_cast<const us8*>(&hld[q * HROW + klo]);
            us8 a1 = *reinterpret_cast<const us8*>(&hld[q * HROW + klo + 16]);
            v16bf af = mk_frag(a0, a1);
            v16bf bf0 = load_b_frag(br0 + kb + half * 8);
            v16bf bf1 = load_b_frag(br1 + kb + half * 8);
            acc0 = wmma_bf16(af, bf0, acc0);
            acc1 = wmma_bf16(af, bf1, acc1);
        }
        __syncthreads();   // everyone done reading h for this step
        if (half == 0) {   // rows m = 0..7 live in lanes 0..15, elems 0..7
            #pragma unroll
            for (int r = 0; r < 8; ++r) {
                const int xi = (t * BATCH + r) * HIDDEN;
                float v0 = tanhf(acc0[r] + bias0 + xproj[xi + n0 + q]);
                float v1 = tanhf(acc1[r] + bias1 + xproj[xi + n1 + q]);
                const unsigned short h0 = f2bf(v0), h1 = f2bf(v1);
                hld[r * HROW + n0 + q] = h0;
                hld[r * HROW + n1 + q] = h1;
                hs_bf[xi + n0 + q] = h0;
                hs_bf[xi + n1 + q] = h1;
            }
        }
        __syncthreads();   // h_new fully published before next step reads
    }
}

// ---------------------------------------------------------------------------
// Kernel D: logits = hs @ W_fc^T + b_fc.  M=2048, K=1024, N=32000.
//   Grid: (500, 16), block 256 (8 waves) computing 128(M) x 64(N).
//   Double-buffered LDS panels filled with global_load_async_to_lds_b128;
//   inner loop is ds_load_b128 fragments + 4 WMMAs per wave per K-chunk.
// ---------------------------------------------------------------------------
__global__ __launch_bounds__(256)
void fc_kernel(const unsigned short* __restrict__ hs_bf,
               const unsigned short* __restrict__ wfc_bf,
               const float* __restrict__ b_fc,
               float* __restrict__ out) {
    __shared__ __align__(16) unsigned short ldsA[2][128 * PROW];
    __shared__ __align__(16) unsigned short ldsB[2][64 * PROW];

    const int tid  = threadIdx.x;
    const int lane = tid & 31;
    const int wave = tid >> 5;
    const int half = lane >> 4;
    const int q    = lane & 15;

    const int mBlock = blockIdx.y * 128;
    const int nBlock = blockIdx.x * 64;

    // staging coordinates: 16B segments, 4 per 32-wide row
    const int rowS = tid >> 2;     // 0..63
    const int part = tid & 3;      // 16B segment within row

    const unsigned short* gA0 = hs_bf + (mBlock + rowS) * HIDDEN + part * 8;
    const unsigned short* gA1 = hs_bf + (mBlock + 64 + rowS) * HIDDEN + part * 8;
    const unsigned short* gB  = wfc_bf + (nBlock + rowS) * HIDDEN + part * 8;

    #define ISSUE_CHUNK(kc, buf)                                               \
        do {                                                                   \
            const int _ko = (kc) * 32;                                         \
            async_cp16(lds_off32(&ldsA[buf][rowS * PROW + part * 8]),          \
                       gA0 + _ko);                                             \
            async_cp16(lds_off32(&ldsA[buf][(64 + rowS) * PROW + part * 8]),   \
                       gA1 + _ko);                                             \
            async_cp16(lds_off32(&ldsB[buf][rowS * PROW + part * 8]),          \
                       gB + _ko);                                              \
        } while (0)

    ISSUE_CHUNK(0, 0);

    v8f acc[4] = {};
    for (int kc = 0; kc < 32; ++kc) {
        const int cur = kc & 1;
        wait_async0();       // my copies for chunk kc are in LDS
        __syncthreads();     // ... and so are everyone else's; prev reads done
        if (kc + 1 < 32) ISSUE_CHUNK(kc + 1, cur ^ 1);

        const unsigned short* As = &ldsA[cur][0];
        const unsigned short* Bs = &ldsB[cur][0];
        us8 a0 = *reinterpret_cast<const us8*>(As + (wave * 16 + q) * PROW + half * 8);
        us8 a1 = *reinterpret_cast<const us8*>(As + (wave * 16 + q) * PROW + half * 8 + 16);
        v16bf af = mk_frag(a0, a1);
        #pragma unroll
        for (int j = 0; j < 4; ++j) {
            v16bf bf = load_b_frag(Bs + (j * 16 + q) * PROW + half * 8);
            acc[j] = wmma_bf16(af, bf, acc[j]);
        }
    }
    #undef ISSUE_CHUNK

    #pragma unroll
    for (int j = 0; j < 4; ++j) {
        const int n = nBlock + j * 16 + q;
        const float bias = b_fc[n];
        #pragma unroll
        for (int r = 0; r < 8; ++r) {
            const int m = mBlock + wave * 16 + half * 8 + r;
            const int t = m >> 3, b = m & 7;
            out[(b * SEQ + t) * OUTV + n] = acc[j][r] + bias;
        }
    }
}

// ---------------------------------------------------------------------------
// Host entry
// ---------------------------------------------------------------------------
extern "C" void kernel_launch(void* const* d_in, const int* in_sizes, int n_in,
                              void* d_out, int out_size, void* d_ws, size_t ws_size,
                              hipStream_t stream) {
    (void)in_sizes; (void)n_in; (void)out_size; (void)ws_size;

    const long long* inputs = (const long long*)d_in[0];
    const float* embed = (const float*)d_in[1];
    const float* W_ih  = (const float*)d_in[2];
    const float* W_hh  = (const float*)d_in[3];
    const float* b_ih  = (const float*)d_in[4];
    const float* b_hh  = (const float*)d_in[5];
    const float* W_fc  = (const float*)d_in[6];
    const float* b_fc  = (const float*)d_in[7];
    float* out = (float*)d_out;

    // workspace layout (bytes)
    char* ws = (char*)d_ws;
    unsigned short* wih_bf = (unsigned short*)(ws + 0);          //  1.0 MB
    unsigned short* whh_bf = (unsigned short*)(ws + 1048576);    //  2.0 MB
    unsigned short* wfc_bf = (unsigned short*)(ws + 3145728);    // 65.5 MB
    float*          xproj  = (float*)        (ws + 68681728);    //  8.4 MB
    unsigned short* hs_bf  = (unsigned short*)(ws + 77070336);   //  4.2 MB

    const int nIH = HIDDEN * EMBED;    // 524288
    const int nHH = HIDDEN * HIDDEN;   // 1048576
    const int nFC = OUTV * HIDDEN;     // 32768000
    cvt_f32_bf16<<<(nIH / 4 + 255) / 256, 256, 0, stream>>>(W_ih, wih_bf, nIH);
    cvt_f32_bf16<<<(nHH / 4 + 255) / 256, 256, 0, stream>>>(W_hh, whh_bf, nHH);
    cvt_f32_bf16<<<(nFC / 4 + 255) / 256, 256, 0, stream>>>(W_fc, wfc_bf, nFC);

    xproj_kernel<<<dim3(HIDDEN / 128, MROWS / 16), 256, 0, stream>>>(
        inputs, embed, wih_bf, b_ih, xproj);

    rnn_kernel<<<1, 1024, 0, stream>>>(xproj, whh_bf, b_hh, hs_bf);

    fc_kernel<<<dim3(OUTV / 64, MROWS / 128), 256, 0, stream>>>(
        hs_bf, wfc_bf, b_fc, out);
}
